// encoder_rnn_42941083025637
// MI455X (gfx1250) — matrix operand order repaired and checked by reading, not compiled, not run on hardware
//
#include <hip/hip_runtime.h>
#include <hip/hip_bf16.h>
#include <cstdint>
#include <cstddef>

// ---- problem constants ----
#define B_   128
#define T_   1024
#define IN_  256
#define H_   512
#define OUT_ 128
#define F1_  1024   // 2H
#define F2_  2048   // 4H
#define EPS_ 1e-5f

#define L0K  768            // layer0 K = IN + H
#define L1K  1536           // layer1 K = 2H + H
#define L0KT 24             // L0K/32
#define L1KT 48             // L1K/32
#define NQW  4              // n-tiles per wave / per block column group
#define L0W  (NQW * L0KT * 512)   // 49152 elems = 96 KB
#define L1W  (NQW * L1KT * 512)   // 98304 elems = 192 KB
#define LDS_BYTES ((L0W + L1W) * 2)  // 294912 B

typedef __bf16 bf16;
typedef __attribute__((ext_vector_type(16))) __bf16 v16bf;
typedef __attribute__((ext_vector_type(8)))  __bf16 v8bf;
typedef __attribute__((ext_vector_type(8)))  float  v8f;

union V16U { v16bf v; v8bf h[2]; };

__device__ __forceinline__ bf16 f2bf(float f) {
    union { float f; unsigned u; } x; x.f = f;
    unsigned r = x.u + 0x7FFFu + ((x.u >> 16) & 1u);   // round-to-nearest-even
    unsigned short hh = (unsigned short)(r >> 16);
    return __builtin_bit_cast(__bf16, hh);
}
__device__ __forceinline__ float bf2f(bf16 b) {
    unsigned short hh = __builtin_bit_cast(unsigned short, b);
    union { unsigned u; float f; } x; x.u = ((unsigned)hh) << 16;
    return x.f;
}

__device__ __forceinline__ v8f wmma_bf16(v16bf a, v16bf b, v8f c) {
    return __builtin_amdgcn_wmma_f32_16x16x32_bf16(
        false, a, false, b, (short)0, c, false, false);
}

// ---------------- device-wide two-phase barrier (all blocks resident) ----------------
__device__ __forceinline__ void grid_sync(int* cnt, volatile int* gen, int nblocks) {
    __syncthreads();
    if (threadIdx.x == 0) {
        __threadfence();
        int g = *gen;
        if (atomicAdd(cnt, 1) == nblocks - 1) {
            *cnt = 0;
            __threadfence();
            atomicAdd((int*)gen, 1);
        } else {
            while (*gen == g) { __builtin_amdgcn_s_sleep(2); }
        }
        __threadfence();
    }
    __syncthreads();
}

// ---------------- prep kernels ----------------
__global__ void init_barrier_kernel(int* bar) {
    if (threadIdx.x < 8) bar[threadIdx.x] = 0;
}

__global__ void cvt_f32_bf16_kernel(const float* __restrict__ s, bf16* __restrict__ d, size_t n) {
    size_t stride = (size_t)gridDim.x * blockDim.x;
    for (size_t i = (size_t)blockIdx.x * blockDim.x + threadIdx.x; i < n; i += stride)
        d[i] = f2bf(s[i]);
}

__global__ void cvt_h0_kernel(const float* __restrict__ h0, bf16* __restrict__ l0, bf16* __restrict__ l1) {
    int i = blockIdx.x * blockDim.x + threadIdx.x;
    if (i >= 4 * B_ * H_) return;
    int dir = i / (B_ * H_);
    int r   = i % (B_ * H_);
    bf16 v = f2bf(h0[i]);
    if (dir < 2) l0[dir * B_ * H_ + r] = v;
    else         l1[(dir - 2) * B_ * H_ + r] = v;
}

__global__ void bias_prep_kernel(const float* bi0, const float* bh0,
                                 const float* bi1, const float* bh1,
                                 float* o0, float* o1) {
    int i = blockIdx.x * blockDim.x + threadIdx.x;
    if (i < 2 * H_) o0[i] = bi0[i] + bh0[i];
    else if (i < 4 * H_) { int j = i - 2 * H_; o1[j] = bi1[j] + bh1[j]; }
}

// Swizzle W[n][k] (row-major, K split across two sources) into WMMA B-operand layout:
// per (ntile, ktile): 32 lanes x 16 contiguous bf16. lane = ((k%32 / 8)&1)*16 + n%16, j = k%8 + (k%32 / 16)*8.
__global__ void swz_kernel(const float* __restrict__ srcA, int KA,
                           const float* __restrict__ srcB, int KB,
                           bf16* __restrict__ dst, int N, int Ktot) {
    size_t total = (size_t)N * Ktot;
    size_t stride = (size_t)gridDim.x * blockDim.x;
    int KT = Ktot >> 5;
    for (size_t i = (size_t)blockIdx.x * blockDim.x + threadIdx.x; i < total; i += stride) {
        int n = (int)(i / Ktot);
        int k = (int)(i % Ktot);
        float v = (k < KA) ? srcA[(size_t)n * KA + k] : srcB[(size_t)n * KB + (k - KA)];
        int ntile = n >> 4, nl = n & 15, ktile = k >> 5, kk = k & 31;
        int lanei = (((kk >> 3) & 1) << 4) | nl;
        int j = (kk & 7) | ((kk >> 4) << 3);
        dst[((((size_t)ntile * KT + ktile) << 5) + lanei) * 16 + j] = f2bf(v);
    }
}

// ---------------- persistent bidirectional 2-layer ReLU-RNN ----------------
// 16 blocks x 256 threads. block = (dir, n-quad): 4 consecutive n-tiles (64 cols).
// wave = m-tile (8 waves = all 128 batch rows). Each wave: 4 accumulators (4 n-tiles),
// A loaded once per K-chunk and reused by 4 WMMAs. Weights live in LDS (288 KB),
// filled once before the scan, re-read every step via ds_load.
__global__ __launch_bounds__(256)
void rnn_persistent_kernel(const bf16* __restrict__ x_bf,
                           const bf16* __restrict__ w0swz,   // [2][32 ntile][24 ktile][32][16]
                           const bf16* __restrict__ w1swz,   // [2][32 ntile][48 ktile][32][16]
                           const float* __restrict__ bias0,  // [2][512]
                           const float* __restrict__ bias1,  // [2][512]
                           bf16* h0a, bf16* h0b,             // layer0 state [2][B][H] x2 buffers
                           bf16* h1a, bf16* h1b,             // layer1 state
                           int* bar) {
    extern __shared__ __bf16 ldsw[];   // [L0W + L1W]

    const int tid  = threadIdx.x;
    const int wv   = tid >> 5;
    const int lane = tid & 31;
    const int nb   = gridDim.x;
    int* bcnt = bar;
    volatile int* bgen = (volatile int*)(bar + 1);

    const int d  = blockIdx.x >> 3;   // direction 0/1
    const int nq = blockIdx.x & 7;    // n-quad: ntiles nq*4 .. nq*4+3

    // ---- one-time LDS weight fill (contiguous slices of the swizzled layout) ----
    {
        const bf16* g0 = w0swz + (size_t)d * (L0K * H_) + (size_t)(nq * NQW) * L0KT * 512;
        const bf16* g1 = w1swz + (size_t)d * (L1K * H_) + (size_t)(nq * NQW) * L1KT * 512;
        for (int i = tid * 8; i < L0W; i += 256 * 8)
            *(v8bf*)(&ldsw[i]) = *(const v8bf*)(g0 + i);
        for (int i = tid * 8; i < L1W; i += 256 * 8)
            *(v8bf*)(&ldsw[L0W + i]) = *(const v8bf*)(g1 + i);
        __syncthreads();
    }

    const int nl    = lane & 15;
    const int m     = wv * 16 + nl;              // A-matrix row
    const int koff  = (lane >> 4) << 3;          // K sub-offset per A layout half
    const int mb    = wv * 16 + ((lane >> 4) << 3);
    const int nbase = nq * 64;

    float bv0[NQW], bv1[NQW];
    #pragma unroll
    for (int c = 0; c < NQW; ++c) {
        bv0[c] = bias0[d * H_ + nbase + c * 16 + nl];
        bv1[c] = bias1[d * H_ + nbase + c * 16 + nl];
    }

    const size_t mo = (size_t)m * H_;

    for (int t = 0; t < T_; ++t) {
        bf16* h0prev = (t & 1) ? h0b : h0a;
        bf16* h0next = (t & 1) ? h0a : h0b;
        bf16* h1prev = (t & 1) ? h1b : h1a;
        bf16* h1next = (t & 1) ? h1a : h1b;

        // ---- layer 0: [x_t | h0prev[d]] (K=768) x Wcat0[d] ----
        {
            v8f acc[NQW];
            #pragma unroll
            for (int c = 0; c < NQW; ++c) {
                #pragma unroll
                for (int r = 0; r < 8; ++r) acc[c][r] = bv0[c];
            }
            const bf16* xrow = x_bf + ((size_t)m * T_ + t) * IN_;
            const bf16* hrow = h0prev + (size_t)d * (B_ * H_) + mo;
            __builtin_prefetch(xrow + IN_, 0, 0);   // next timestep's x row

            // segment 1: K 0..255 from x_t (no selects: fixed base pointer)
            #pragma unroll
            for (int kt = 0; kt < 8; ++kt) {
                const int kb = kt << 5;
                V16U a;
                a.h[0] = *(const v8bf*)(xrow + kb + koff);
                a.h[1] = *(const v8bf*)(xrow + kb + 16 + koff);
                #pragma unroll
                for (int c = 0; c < NQW; ++c) {
                    v16bf bm = *(const v16bf*)(&ldsw[((c * L0KT + kt) * 32 + lane) * 16]);
                    acc[c] = wmma_bf16(a.v, bm, acc[c]);
                }
            }
            // segment 2: K 256..767 from h0prev
            #pragma unroll 4
            for (int kt = 8; kt < L0KT; ++kt) {
                const int kb = (kt << 5) - IN_;
                V16U a;
                a.h[0] = *(const v8bf*)(hrow + kb + koff);
                a.h[1] = *(const v8bf*)(hrow + kb + 16 + koff);
                #pragma unroll
                for (int c = 0; c < NQW; ++c) {
                    v16bf bm = *(const v16bf*)(&ldsw[((c * L0KT + kt) * 32 + lane) * 16]);
                    acc[c] = wmma_bf16(a.v, bm, acc[c]);
                }
            }
            bf16* outp = h0next + (size_t)d * (B_ * H_);
            #pragma unroll
            for (int c = 0; c < NQW; ++c) {
                const int n = nbase + c * 16 + nl;
                #pragma unroll
                for (int r = 0; r < 8; ++r)
                    outp[(size_t)(mb + r) * H_ + n] = f2bf(fmaxf(acc[c][r], 0.0f));
            }
        }
        grid_sync(bcnt, bgen, nb);

        // ---- layer 1: [h0_fwd | h0_bwd | h1prev[d]] (K=1536) x Wcat1[d] ----
        {
            v8f acc[NQW];
            #pragma unroll
            for (int c = 0; c < NQW; ++c) {
                #pragma unroll
                for (int r = 0; r < 8; ++r) acc[c][r] = bv1[c];
            }
            const bf16* h0f  = h0next + mo;
            const bf16* h0bd = h0next + B_ * H_ + mo;
            const bf16* h1r  = h1prev + (size_t)d * (B_ * H_) + mo;
            const bf16* lw1  = ldsw + L0W;

            // segment 1: K 0..511 from h0 forward
            #pragma unroll 4
            for (int kt = 0; kt < 16; ++kt) {
                const int kb = kt << 5;
                V16U a;
                a.h[0] = *(const v8bf*)(h0f + kb + koff);
                a.h[1] = *(const v8bf*)(h0f + kb + 16 + koff);
                #pragma unroll
                for (int c = 0; c < NQW; ++c) {
                    v16bf bm = *(const v16bf*)(lw1 + ((c * L1KT + kt) * 32 + lane) * 16);
                    acc[c] = wmma_bf16(a.v, bm, acc[c]);
                }
            }
            // segment 2: K 512..1023 from h0 backward
            #pragma unroll 4
            for (int kt = 16; kt < 32; ++kt) {
                const int kb = (kt << 5) - H_;
                V16U a;
                a.h[0] = *(const v8bf*)(h0bd + kb + koff);
                a.h[1] = *(const v8bf*)(h0bd + kb + 16 + koff);
                #pragma unroll
                for (int c = 0; c < NQW; ++c) {
                    v16bf bm = *(const v16bf*)(lw1 + ((c * L1KT + kt) * 32 + lane) * 16);
                    acc[c] = wmma_bf16(a.v, bm, acc[c]);
                }
            }
            // segment 3: K 1024..1535 from h1prev
            #pragma unroll 4
            for (int kt = 32; kt < L1KT; ++kt) {
                const int kb = (kt << 5) - 2 * H_;
                V16U a;
                a.h[0] = *(const v8bf*)(h1r + kb + koff);
                a.h[1] = *(const v8bf*)(h1r + kb + 16 + koff);
                #pragma unroll
                for (int c = 0; c < NQW; ++c) {
                    v16bf bm = *(const v16bf*)(lw1 + ((c * L1KT + kt) * 32 + lane) * 16);
                    acc[c] = wmma_bf16(a.v, bm, acc[c]);
                }
            }
            bf16* outp = h1next + (size_t)d * (B_ * H_);
            #pragma unroll
            for (int c = 0; c < NQW; ++c) {
                const int n = nbase + c * 16 + nl;
                #pragma unroll
                for (int r = 0; r < 8; ++r)
                    outp[(size_t)(mb + r) * H_ + n] = f2bf(fmaxf(acc[c][r], 0.0f));
            }
        }
        grid_sync(bcnt, bgen, nb);
    }
}

// ---------------- head ----------------
__global__ void bn1_kernel(const bf16* __restrict__ h1fin,
                           const float* __restrict__ g, const float* __restrict__ bb,
                           bf16* __restrict__ a1) {
    int f = blockIdx.x * blockDim.x + threadIdx.x;
    if (f >= F1_) return;
    int dir = f >> 9, ff = f & 511;
    const bf16* p = h1fin + (size_t)dir * B_ * H_ + ff;
    float s = 0.f, s2 = 0.f;
    for (int b = 0; b < B_; ++b) { float v = bf2f(p[(size_t)b * H_]); s += v; s2 += v * v; }
    float mean = s / B_;
    float var  = s2 / B_ - mean * mean;
    float isd  = rsqrtf(var + EPS_);
    float gg = g[f] * isd;
    float cc = bb[f] - mean * gg;
    for (int b = 0; b < B_; ++b) {
        float v = bf2f(p[(size_t)b * H_]);
        a1[(size_t)b * F1_ + f] = f2bf(fmaxf(v * gg + cc, 0.0f));
    }
}

__global__ void bn2_kernel(const float* __restrict__ z,
                           const float* __restrict__ g, const float* __restrict__ bb,
                           bf16* __restrict__ a2) {
    int f = blockIdx.x * blockDim.x + threadIdx.x;
    if (f >= F2_) return;
    const float* p = z + f;
    float s = 0.f, s2 = 0.f;
    for (int b = 0; b < B_; ++b) { float v = p[(size_t)b * F2_]; s += v; s2 += v * v; }
    float mean = s / B_;
    float var  = s2 / B_ - mean * mean;
    float isd  = rsqrtf(var + EPS_);
    float gg = g[f] * isd;
    float cc = bb[f] - mean * gg;
    for (int b = 0; b < B_; ++b)
        a2[(size_t)b * F2_ + f] = f2bf(fmaxf(p[(size_t)b * F2_] * gg + cc, 0.0f));
}

// Generic WMMA head GEMM: out[M=128,N] = act(A[bf16 M,K] * Bswz + bias). act: 0=none, 1=sigmoid.
__global__ __launch_bounds__(256)
void gemm_head_kernel(const bf16* __restrict__ A, const bf16* __restrict__ Bswz,
                      const float* __restrict__ bias, float* __restrict__ out,
                      int K, int N, int act) {
    const int wtile = blockIdx.x * 8 + (threadIdx.x >> 5);
    const int lane  = threadIdx.x & 31;
    const int ntiles = N >> 4;
    const int total  = (B_ >> 4) * ntiles;
    if (wtile >= total) return;
    const int mt = wtile / ntiles, nt = wtile % ntiles;
    const int KT = K >> 5;
    const int nl = lane & 15;
    const int m  = mt * 16 + nl;
    const int koff = (lane >> 4) << 3;
    const float bv = bias[nt * 16 + nl];
    v8f acc;
    #pragma unroll
    for (int r = 0; r < 8; ++r) acc[r] = bv;
    const bf16* arow = A + (size_t)m * K;
    const bf16* brow = Bswz + (size_t)nt * KT * 512;
    #pragma unroll 4
    for (int kt = 0; kt < KT; ++kt) {
        const int kb = kt << 5;
        V16U a;
        a.h[0] = *(const v8bf*)(arow + kb + koff);
        a.h[1] = *(const v8bf*)(arow + kb + 16 + koff);
        v16bf bm = *(const v16bf*)(brow + ((size_t)(kb + lane)) * 16);
        acc = wmma_bf16(a.v, bm, acc);
    }
    const int mbase = mt * 16 + ((lane >> 4) << 3);
    const int n = nt * 16 + nl;
    #pragma unroll
    for (int r = 0; r < 8; ++r) {
        float v = acc[r];
        if (act == 1) v = 1.0f / (1.0f + expf(-v));
        out[(size_t)(mbase + r) * N + n] = v;
    }
}

// ---------------- host-side orchestration ----------------
extern "C" void kernel_launch(void* const* d_in, const int* in_sizes, int n_in,
                              void* d_out, int out_size, void* d_ws, size_t ws_size,
                              hipStream_t stream) {
    const float* x     = (const float*)d_in[0];
    const float* h0    = (const float*)d_in[1];
    const float* W_ih0 = (const float*)d_in[2];
    const float* W_hh0 = (const float*)d_in[3];
    const float* b_ih0 = (const float*)d_in[4];
    const float* b_hh0 = (const float*)d_in[5];
    const float* W_ih1 = (const float*)d_in[6];
    const float* W_hh1 = (const float*)d_in[7];
    const float* b_ih1 = (const float*)d_in[8];
    const float* b_hh1 = (const float*)d_in[9];
    const float* bn1_g = (const float*)d_in[10];
    const float* bn1_b = (const float*)d_in[11];
    const float* fc1_W = (const float*)d_in[12];
    const float* fc1_b = (const float*)d_in[13];
    const float* bn2_g = (const float*)d_in[14];
    const float* bn2_b = (const float*)d_in[15];
    const float* fc2_W = (const float*)d_in[16];
    const float* fc2_b = (const float*)d_in[17];

    uint8_t* p = (uint8_t*)d_ws;
    auto alloc = [&](size_t bytes) -> void* {
        void* r = (void*)p;
        p += (bytes + 255) & ~(size_t)255;
        return r;
    };
    bf16* x_bf   = (bf16*)alloc((size_t)B_ * T_ * IN_ * sizeof(bf16));     // 64 MB
    bf16* w0swz  = (bf16*)alloc((size_t)2 * L0K * H_ * sizeof(bf16));      // 1.5 MB
    bf16* w1swz  = (bf16*)alloc((size_t)2 * L1K * H_ * sizeof(bf16));      // 3 MB
    bf16* fc1swz = (bf16*)alloc((size_t)F1_ * F2_ * sizeof(bf16));         // 4 MB
    bf16* fc2swz = (bf16*)alloc((size_t)F2_ * OUT_ * sizeof(bf16));        // 0.5 MB
    bf16* h0a    = (bf16*)alloc((size_t)2 * B_ * H_ * sizeof(bf16));
    bf16* h0b    = (bf16*)alloc((size_t)2 * B_ * H_ * sizeof(bf16));
    bf16* h1a    = (bf16*)alloc((size_t)2 * B_ * H_ * sizeof(bf16));
    bf16* h1b    = (bf16*)alloc((size_t)2 * B_ * H_ * sizeof(bf16));
    float* bias0 = (float*)alloc((size_t)2 * H_ * sizeof(float));
    float* bias1 = (float*)alloc((size_t)2 * H_ * sizeof(float));
    bf16* a1     = (bf16*)alloc((size_t)B_ * F1_ * sizeof(bf16));
    float* z1    = (float*)alloc((size_t)B_ * F2_ * sizeof(float));
    bf16* a2     = (bf16*)alloc((size_t)B_ * F2_ * sizeof(bf16));
    int*  bar    = (int*)alloc(256);

    // ---- prep ----
    init_barrier_kernel<<<1, 32, 0, stream>>>(bar);
    cvt_f32_bf16_kernel<<<2048, 256, 0, stream>>>(x, x_bf, (size_t)B_ * T_ * IN_);
    cvt_h0_kernel<<<(4 * B_ * H_ + 255) / 256, 256, 0, stream>>>(h0, h0a, h1a);
    bias_prep_kernel<<<8, 256, 0, stream>>>(b_ih0, b_hh0, b_ih1, b_hh1, bias0, bias1);
    for (int d = 0; d < 2; ++d)
        swz_kernel<<<512, 256, 0, stream>>>(W_ih0 + (size_t)d * H_ * IN_, IN_,
                                            W_hh0 + (size_t)d * H_ * H_,  H_,
                                            w0swz + (size_t)d * L0K * H_, H_, L0K);
    for (int d = 0; d < 2; ++d)
        swz_kernel<<<512, 256, 0, stream>>>(W_ih1 + (size_t)d * H_ * F1_, F1_,
                                            W_hh1 + (size_t)d * H_ * H_,  H_,
                                            w1swz + (size_t)d * L1K * H_, H_, L1K);
    swz_kernel<<<512, 256, 0, stream>>>(fc1_W, F1_, fc1_W, 0, fc1swz, F2_, F1_);
    swz_kernel<<<512, 256, 0, stream>>>(fc2_W, F2_, fc2_W, 0, fc2swz, OUT_, F2_);

    // ---- persistent RNN scan: 16 blocks x 8 waves, 288 KB LDS weight cache ----
    rnn_persistent_kernel<<<16, 256, LDS_BYTES, stream>>>(x_bf, w0swz, w1swz, bias0, bias1,
                                                          h0a, h0b, h1a, h1b, bar);

    // final layer-1 state after t=1023 lives in buffer A (parity (1023+1)&1 == 0)
    bn1_kernel<<<(F1_ + 255) / 256, 256, 0, stream>>>(h1a, bn1_g, bn1_b, a1);
    gemm_head_kernel<<<(8 * (F2_ / 16)) / 8, 256, 0, stream>>>(a1, fc1swz, fc1_b, z1, F1_, F2_, 0);
    bn2_kernel<<<(F2_ + 255) / 256, 256, 0, stream>>>(z1, bn2_g, bn2_b, a2);
    gemm_head_kernel<<<(8 * (OUT_ / 16)) / 8, 256, 0, stream>>>(a2, fc2swz, fc2_b, (float*)d_out, F2_, OUT_, 1);

    (void)in_sizes; (void)n_in; (void)out_size; (void)ws_size;
}
